// FourierWindowAttention_71296457114376
// MI455X (gfx1250) — matrix operand order repaired and checked by reading, not compiled, not run on hardware
//
#include <hip/hip_runtime.h>

typedef __attribute__((ext_vector_type(16))) _Float16 v16h;
typedef __attribute__((ext_vector_type(8)))  float    v8f;

#define N_TOK 64
#define DIMC  256
#define NHEAD 8
#define HDIM  32
#define NF    40
#define NFP   48

// ---- LDS byte offsets (phase-ordered region reuse, 288KB high-water) ----
#define OFF_XS_F32   0        // [64][256] f32 (ph1-2); per-wave P scratch 8KB*8 (ph4,6); x_f accum f32 (ph7)
#define OFF_XS_BF    65536    // [64][256] f16 (ph1,3); XSUM f16 (ph7b,8)
#define OFF_XF_BF    98304    // [96][256] f16 fft re/im (ph2,5); xf_res re/im f16 [48][256]x2 (ph6,7)
#define OFF_QKV      147456   // [96][768] f16 qkv (spatial rows 0..63 in ph3-4; fourier 0..95 in ph5-6)
#define LDS_BYTES    327680   // request full 320KB so padded-K reads stay in-range

// ============================ WMMA tile helpers ============================
// A 16x32 f16 tile: lane = M (0..15), half (lane>=16) selects K sub-blocks.
__device__ static inline v16h ld_a16(const _Float16* p, int row0, int stride, int k0, int lane) {
  const int m = lane & 15, half = lane >> 4;
  union { v16h v; unsigned u[8]; } t;
  const _Float16* base = p + (row0 + m) * stride + k0;
#pragma unroll
  for (int r = 0; r < 8; ++r) {
    const int kb = (half << 4) + (r << 1);
    t.u[r] = *(const unsigned*)(base + kb);
  }
  return t.v;
}

// B 32x16 tile where memory holds B^T (rows = N, contiguous in K): element(k,n)=p[(row0+n)*stride+k0+k]
__device__ static inline v16h ld_bT16(const _Float16* p, int row0, int stride, int k0, int lane) {
  const int n = lane & 15, kh = (lane >> 4) << 4;
  union { v16h v; unsigned u[8]; } t;
  const _Float16* base = p + (row0 + n) * stride + k0 + kh;
#pragma unroll
  for (int r = 0; r < 8; ++r) t.u[r] = *(const unsigned*)(base + (r << 1));
  return t.v;
}

// B 32x16 tile from row-major K x N storage: element(k,n)=p[(k0+k)*stride+col0+n]
__device__ static inline v16h ld_bRM16(const _Float16* p, int k0, int stride, int col0, int lane) {
  const int n = lane & 15, kh = (lane >> 4) << 4;
  union { v16h v; _Float16 h[16]; } t;
  const _Float16* base = p + (k0 + kh) * stride + col0 + n;
#pragma unroll
  for (int r = 0; r < 8; ++r) {
    t.h[2 * r]     = base[(2 * r) * stride];
    t.h[2 * r + 1] = base[(2 * r + 1) * stride];
  }
  return t.v;
}

__device__ static inline v8f wmma16(v16h a, v16h b, v8f c) {
  return __builtin_amdgcn_wmma_f32_16x16x32_f16(false, a, false, b, (short)0, c, false, false);
}

// reductions within each 16-lane half (row of a C/D tile lives in one half)
__device__ static inline float rmax16(float v) {
#pragma unroll
  for (int m = 1; m < 16; m <<= 1) v = fmaxf(v, __shfl_xor(v, m, 32));
  return v;
}
__device__ static inline float rsum16(float v) {
#pragma unroll
  for (int m = 1; m < 16; m <<= 1) v += __shfl_xor(v, m, 32);
  return v;
}

// qkv GEMM: D[rowTiles*16][768] = A[rowTiles*16][256] @ W^T + b   (W stored [768][256])
__device__ static inline void gemm_qkv(const _Float16* A, int rowTiles,
                                       const _Float16* W, const float* bvec,
                                       _Float16* D, int wave, int lane) {
  const int total = rowTiles * 48;
  for (int t = wave; t < total; t += 8) {
    const int rt = t / 48, ct = t - rt * 48;
    v8f acc = {};
#pragma unroll
    for (int k0 = 0; k0 < 256; k0 += 32) {
      v16h a  = ld_a16(A, rt * 16, 256, k0, lane);
      v16h bm = ld_bT16(W, ct * 16, 256, k0, lane);
      acc = wmma16(a, bm, acc);
    }
    const float bj = bvec[ct * 16 + (lane & 15)];
    const int rh = (lane >> 4) << 3, nn = lane & 15;
#pragma unroll
    for (int r = 0; r < 8; ++r)
      D[(rt * 16 + rh + r) * 768 + ct * 16 + nn] = (_Float16)(acc[r] + bj);
  }
}

// ============================ prep: weights fp32 -> fp16 ============================
__global__ void fwa_prep(const float* __restrict__ qkv_w, const float* __restrict__ proj_w,
                         _Float16* __restrict__ wq, _Float16* __restrict__ wp) {
  const int i = blockIdx.x * blockDim.x + threadIdx.x;
  if (i < 768 * 256) wq[i] = (_Float16)qkv_w[i];
  if (i < 256 * 256) wp[i] = (_Float16)proj_w[i];
}

// ============================ main fused kernel ============================
__global__ __launch_bounds__(256)
void fwa_main(const float* __restrict__ x, const float* __restrict__ bias_table,
              const float* __restrict__ qkv_b, const float* __restrict__ proj_b,
              const _Float16* __restrict__ wq, const _Float16* __restrict__ wp,
              float* __restrict__ out, float* __restrict__ attn_out) {
  extern __shared__ char smem[];
  const int tid = threadIdx.x, lane = tid & 31, wave = tid >> 5;
  const int b = blockIdx.x;
  const int h = wave;  // one head per wave
  const float scale = 0.17677669529663689f;  // 1/sqrt(32)
  const float C8[8] = {1.f, 0.70710678118654752f, 0.f, -0.70710678118654752f,
                       -1.f, -0.70710678118654752f, 0.f, 0.70710678118654752f};
  const float S8[8] = {0.f, 0.70710678118654752f, 1.f, 0.70710678118654752f,
                       0.f, -0.70710678118654752f, -1.f, -0.70710678118654752f};

  float*     XS   = (float*)(smem + OFF_XS_F32);
  _Float16*  XSB  = (_Float16*)(smem + OFF_XS_BF);
  _Float16*  XFB  = (_Float16*)(smem + OFF_XF_BF);   // fft tokens [96][256]
  _Float16*  QKV  = (_Float16*)(smem + OFF_QKV);     // [96][768]

  __builtin_prefetch((const void*)wq, 0, 1);

  // ---------------- phase 1: load x, make f16 copy, zero fft pad rows ----------------
  const float* xb = x + (size_t)b * (N_TOK * DIMC);
  for (int i = tid; i < N_TOK * DIMC / 4; i += 256) {
    float4 v = ((const float4*)xb)[i];
    ((float4*)XS)[i] = v;
    XSB[4 * i + 0] = (_Float16)v.x;
    XSB[4 * i + 1] = (_Float16)v.y;
    XSB[4 * i + 2] = (_Float16)v.z;
    XSB[4 * i + 3] = (_Float16)v.w;
  }
  {  // zero pad rows 40..47 and 88..95 of XFB (128 u32 per row)
    unsigned* XFu = (unsigned*)XFB;
    for (int i = tid; i < 2048; i += 256) {
      const int row  = (i < 1024) ? (40 + (i >> 7)) : (88 + ((i - 1024) >> 7));
      XFu[row * 128 + (i & 127)] = 0u;
    }
  }
  __syncthreads();

  // ---------------- phase 2: per-channel 8x8 rFFT (ortho), write f16 tokens ----------------
  {
    const int c = tid;  // 256 threads == 256 channels
    float xr[8][8];
#pragma unroll
    for (int n = 0; n < 64; ++n) xr[n >> 3][n & 7] = XS[n * 256 + c];
    float rr[8][5], ri[8][5];
#pragma unroll
    for (int r = 0; r < 8; ++r)
#pragma unroll
      for (int kw = 0; kw < 5; ++kw) {
        float a = 0.f, bb = 0.f;
#pragma unroll
        for (int w = 0; w < 8; ++w) {
          const int t = (kw * w) & 7;
          a += xr[r][w] * C8[t];
          bb -= xr[r][w] * S8[t];
        }
        rr[r][kw] = a; ri[r][kw] = bb;
      }
#pragma unroll
    for (int kh = 0; kh < 8; ++kh)
#pragma unroll
      for (int kw = 0; kw < 5; ++kw) {
        float a = 0.f, bb = 0.f;
#pragma unroll
        for (int r = 0; r < 8; ++r) {
          const int t = (kh * r) & 7;
          a  += rr[r][kw] * C8[t] + ri[r][kw] * S8[t];
          bb += ri[r][kw] * C8[t] - rr[r][kw] * S8[t];
        }
        const int row = kh * 5 + kw;
        XFB[row * 256 + c]        = (_Float16)(a * 0.125f);
        XFB[(48 + row) * 256 + c] = (_Float16)(bb * 0.125f);
      }
  }
  __syncthreads();

  // ---------------- phase 3: spatial qkv GEMM (WMMA) ----------------
  gemm_qkv(XSB, 4, wq, qkv_b, QKV, wave, lane);
  __syncthreads();

  // ---------------- phase 4: spatial attention (one head per wave) ----------------
  v8f Osp[4][2];  // spatial head output, lives in VGPRs until phase 7b
  {
    _Float16* P = (_Float16*)(smem + OFF_XS_F32) + wave * 4096;  // [64][64] f16 per wave
    v8f S[4][4];
#pragma unroll
    for (int nt = 0; nt < 4; ++nt) {
      v16h qa = ld_a16(QKV, nt * 16, 768, h * 32, lane);
#pragma unroll
      for (int mt = 0; mt < 4; ++mt) {
        v16h kb = ld_bT16(QKV, mt * 16, 768, 256 + h * 32, lane);
        v8f z = {};
        S[nt][mt] = wmma16(qa, kb, z);
      }
    }
    float* attn_b = attn_out + (((size_t)b * NHEAD + h) * 64) * 64;
    const int rh = (lane >> 4) << 3, nn = lane & 15;
#pragma unroll
    for (int nt = 0; nt < 4; ++nt) {
#pragma unroll
      for (int r = 0; r < 8; ++r) {
        const int n = nt * 16 + rh + r;
        float s[4];
        float mx = -1e30f;
#pragma unroll
        for (int mt = 0; mt < 4; ++mt) {
          const int m = mt * 16 + nn;
          const int dy = (n >> 3) - (m >> 3) + 7;
          const int dx = (n & 7) - (m & 7) + 7;
          const float bv = bias_table[(dy * 15 + dx) * 8 + h];
          s[mt] = S[nt][mt][r] * scale + bv;
          mx = fmaxf(mx, s[mt]);
        }
        mx = rmax16(mx);
        float es = 0.f;
#pragma unroll
        for (int mt = 0; mt < 4; ++mt) { s[mt] = __expf(s[mt] - mx); es += s[mt]; }
        es = rsum16(es);
        const float inv = 1.0f / es;
#pragma unroll
        for (int mt = 0; mt < 4; ++mt) {
          const int m = mt * 16 + nn;
          const float p = s[mt] * inv;
          attn_b[n * 64 + m] = p;
          P[n * 64 + m] = (_Float16)p;
        }
      }
    }
    // O = P @ V
#pragma unroll
    for (int nt = 0; nt < 4; ++nt)
#pragma unroll
      for (int ct = 0; ct < 2; ++ct) {
        v8f acc = {};
#pragma unroll
        for (int kt = 0; kt < 2; ++kt) {
          v16h pa = ld_a16(P, nt * 16, 64, kt * 32, lane);
          v16h vb = ld_bRM16(QKV, kt * 32, 768, 512 + h * 32 + ct * 16, lane);
          acc = wmma16(pa, vb, acc);
        }
        Osp[nt][ct] = acc;
      }
  }
  __syncthreads();

  // ---------------- phase 5: fourier qkv GEMM (96 padded rows) ----------------
  gemm_qkv(XFB, 6, wq, qkv_b, QKV, wave, lane);
  __syncthreads();

  // ---------------- phase 6: fourier complex attention ----------------
  {
    _Float16* Pf = (_Float16*)(smem + OFF_XS_F32) + wave * 4096;  // [48][64] f16 per wave
    for (int i = lane; i < 48 * 16; i += 32) {                    // zero pad cols 48..63
      Pf[(i >> 4) * 64 + 48 + (i & 15)] = (_Float16)0.f;
    }
    const _Float16* QFr = QKV;             // rows 0..47: real tokens
    const _Float16* QFi = QKV + 48 * 768;  // rows 48..95: imag tokens
    v8f Sr[3][3], Si[3][3];
#pragma unroll
    for (int nt = 0; nt < 3; ++nt) {
      v16h qr = ld_a16(QFr, nt * 16, 768, h * 32, lane);
      v16h qi = ld_a16(QFi, nt * 16, 768, h * 32, lane);
#pragma unroll
      for (int mt = 0; mt < 3; ++mt) {
        v16h kr = ld_bT16(QFr, mt * 16, 768, 256 + h * 32, lane);
        v16h ki = ld_bT16(QFi, mt * 16, 768, 256 + h * 32, lane);
        v8f z = {};
        v8f t_rr = wmma16(qr, kr, z);
        v8f t_ii = wmma16(qi, ki, z);
        v8f t_ri = wmma16(qr, ki, z);
        Sr[nt][mt] = t_rr - t_ii;            // ISA: f16 WMMA has no A/B negate -> VALU sub
        Si[nt][mt] = wmma16(qi, kr, t_ri);   // accumulate via C operand
      }
    }
    const int rh = (lane >> 4) << 3, nn = lane & 15;
#pragma unroll
    for (int nt = 0; nt < 3; ++nt) {
#pragma unroll
      for (int r = 0; r < 8; ++r) {
        float s[3];
        float mx = -1e30f;
#pragma unroll
        for (int mt = 0; mt < 3; ++mt) {
          const int m = mt * 16 + nn;
          const float a = Sr[nt][mt][r], bb = Si[nt][mt][r];
          const float mag = scale * __builtin_sqrtf(a * a + bb * bb);
          s[mt] = (m < NF) ? mag : -1e30f;
          mx = fmaxf(mx, s[mt]);
        }
        mx = rmax16(mx);
        float es = 0.f;
#pragma unroll
        for (int mt = 0; mt < 3; ++mt) { s[mt] = __expf(s[mt] - mx); es += s[mt]; }
        es = rsum16(es);
        const float inv = 1.0f / es;
        const int n = nt * 16 + rh + r;
#pragma unroll
        for (int mt = 0; mt < 3; ++mt)
          Pf[n * 64 + mt * 16 + nn] = (_Float16)(s[mt] * inv);
      }
    }
    // xf_res = P @ (Vr + i Vi)  -> f16 [48][256] re/im reusing XFB region
    _Float16* XFR = XFB;
    _Float16* XFI = XFB + 48 * 256;
#pragma unroll
    for (int nt = 0; nt < 3; ++nt)
#pragma unroll
      for (int ct = 0; ct < 2; ++ct) {
        v8f ar = {}, ai = {};
#pragma unroll
        for (int kt = 0; kt < 2; ++kt) {
          v16h pa = ld_a16(Pf, nt * 16, 64, kt * 32, lane);
          v16h vr = ld_bRM16(QFr, kt * 32, 768, 512 + h * 32 + ct * 16, lane);
          v16h vi = ld_bRM16(QFi, kt * 32, 768, 512 + h * 32 + ct * 16, lane);
          ar = wmma16(pa, vr, ar);
          ai = wmma16(pa, vi, ai);
        }
#pragma unroll
        for (int r = 0; r < 8; ++r) {
          const int row = nt * 16 + rh + r;
          XFR[row * 256 + h * 32 + ct * 16 + nn] = (_Float16)ar[r];
          XFI[row * 256 + h * 32 + ct * 16 + nn] = (_Float16)ai[r];
        }
      }
  }
  __syncthreads();

  // ---------------- phase 7: per-channel irFFT (ortho) -> x_f accumulator ----------------
  {
    const int c = tid;
    const _Float16* XFR = XFB;
    const _Float16* XFI = XFB + 48 * 256;
    float Fr[8][5], Fi[8][5];
#pragma unroll
    for (int kh = 0; kh < 8; ++kh)
#pragma unroll
      for (int kw = 0; kw < 5; ++kw) {
        const int row = kh * 5 + kw;
        Fr[kh][kw] = (float)XFR[row * 256 + c];
        Fi[kh][kw] = (float)XFI[row * 256 + c];
      }
    float yr[8][5], yi[8][5];
#pragma unroll
    for (int hh = 0; hh < 8; ++hh)
#pragma unroll
      for (int kw = 0; kw < 5; ++kw) {
        float a = 0.f, bb = 0.f;
#pragma unroll
        for (int kh = 0; kh < 8; ++kh) {
          const int t = (kh * hh) & 7;
          a  += Fr[kh][kw] * C8[t] - Fi[kh][kw] * S8[t];
          bb += Fr[kh][kw] * S8[t] + Fi[kh][kw] * C8[t];
        }
        yr[hh][kw] = a; yi[hh][kw] = bb;
      }
    float* XA = (float*)(smem + OFF_XS_F32);
#pragma unroll
    for (int hh = 0; hh < 8; ++hh)
#pragma unroll
      for (int w = 0; w < 8; ++w) {
        float v = yr[hh][0] + ((w & 1) ? -yr[hh][4] : yr[hh][4]);
#pragma unroll
        for (int kw = 1; kw < 4; ++kw) {
          const int t = (kw * w) & 7;
          v += 2.f * (yr[hh][kw] * C8[t] - yi[hh][kw] * S8[t]);
        }
        XA[(hh * 8 + w) * 256 + c] = v * 0.125f;
      }
  }
  __syncthreads();

  // ---------------- phase 7b: merge spatial (VGPR) + fourier (LDS) -> f16 ----------------
  {
    const float* XA = (const float*)(smem + OFF_XS_F32);
    _Float16* XSUM = (_Float16*)(smem + OFF_XS_BF);
    const int rh = (lane >> 4) << 3, nn = lane & 15;
#pragma unroll
    for (int nt = 0; nt < 4; ++nt)
#pragma unroll
      for (int ct = 0; ct < 2; ++ct)
#pragma unroll
        for (int r = 0; r < 8; ++r) {
          const int n = nt * 16 + rh + r;
          const int cc = h * 32 + ct * 16 + nn;
          XSUM[n * 256 + cc] = (_Float16)(Osp[nt][ct][r] + XA[n * 256 + cc]);
        }
  }
  __syncthreads();

  // ---------------- phase 8: output projection (WMMA) ----------------
  {
    const _Float16* XSUM = (const _Float16*)(smem + OFF_XS_BF);
    float* outb = out + (size_t)b * (N_TOK * DIMC);
    for (int t = wave; t < 64; t += 8) {
      const int rt = t >> 4, ct = t & 15;
      v8f acc = {};
#pragma unroll
      for (int k0 = 0; k0 < 256; k0 += 32) {
        v16h a  = ld_a16(XSUM, rt * 16, 256, k0, lane);
        v16h bm = ld_bT16(wp, ct * 16, 256, k0, lane);
        acc = wmma16(a, bm, acc);
      }
      const float bj = proj_b[ct * 16 + (lane & 15)];
      const int rh = (lane >> 4) << 3, nn = lane & 15;
#pragma unroll
      for (int r = 0; r < 8; ++r)
        outb[(rt * 16 + rh + r) * 256 + ct * 16 + nn] = acc[r] + bj;
    }
  }
}

extern "C" void kernel_launch(void* const* d_in, const int* in_sizes, int n_in,
                              void* d_out, int out_size, void* d_ws, size_t ws_size,
                              hipStream_t stream) {
  const float* x          = (const float*)d_in[0];
  const float* bias_table = (const float*)d_in[1];
  const float* qkv_w      = (const float*)d_in[2];
  const float* qkv_b      = (const float*)d_in[3];
  const float* proj_w     = (const float*)d_in[4];
  const float* proj_b     = (const float*)d_in[5];

  const int B = in_sizes[0] / (N_TOK * DIMC);  // 2048

  _Float16* wq = (_Float16*)d_ws;
  _Float16* wp = (_Float16*)((char*)d_ws + (size_t)768 * 256 * sizeof(_Float16));

  float* out  = (float*)d_out;
  float* attn = out + (size_t)B * N_TOK * DIMC;

  fwa_prep<<<768, 256, 0, stream>>>(qkv_w, proj_w, wq, wp);
  fwa_main<<<B, 256, LDS_BYTES, stream>>>(x, bias_table, qkv_b, proj_b, wq, wp, out, attn);
}
